// MultiHeadAttention_67508295959409
// MI455X (gfx1250) — matrix operand order repaired and checked by reading, not compiled, not run on hardware
//
#include <hip/hip_runtime.h>

// ---------------------------------------------------------------------------
// MI455X (gfx1250) multi-head attention: bf16 WMMA with fp32 accumulation.
//   cvt (f32->bf16 staging) -> proj (Q/K/V) -> flash attention -> out proj
// Wave32, V_WMMA_F32_16X16X32_BF16 everywhere; hot loops are b128 loads + WMMA.
// ---------------------------------------------------------------------------

typedef __attribute__((ext_vector_type(16))) __bf16 v16bf;
typedef __attribute__((ext_vector_type(8)))  float  v8f;

#define D_MODEL   1024
#define NUM_HEADS 16
#define DK        64
#define BATCH     2
#define SEQ       2048
#define MROWS     (BATCH * SEQ)            // 4096
#define XSZ       ((size_t)MROWS * D_MODEL)     // 4 Mi elems (activations)
#define WSZ       ((size_t)D_MODEL * D_MODEL)   // 1 Mi elems (weights)

// float -> bf16 (native cast: RNE; clang picks hw cvt on gfx1250)
__device__ __forceinline__ __bf16 f2bf(float f) { return (__bf16)f; }

union BFrag {
  v16bf v;
  uint4 q[2];
  __bf16 e[16];
};

// A-fragment (16x32 bf16): lane half lo holds K={k0..k0+7, k0+16..k0+23}.
__device__ __forceinline__ v16bf load_a_bf16(const __bf16* row, int k0) {
  BFrag u;
  u.q[0] = *(const uint4*)(row + k0);
  u.q[1] = *(const uint4*)(row + k0 + 16);
  return u.v;
}
// B-fragment (32x16 bf16): lane half lo holds K={k0..k0+7, k0+16..k0+23}.
__device__ __forceinline__ v16bf load_b_bf16(const __bf16* row, int k0) {
  BFrag u;
  u.q[0] = *(const uint4*)(row + k0);
  u.q[1] = *(const uint4*)(row + k0 + 16);
  return u.v;
}

__device__ __forceinline__ v8f wmma_bf16(v16bf a, v16bf b, v8f c) {
  return __builtin_amdgcn_wmma_f32_16x16x32_bf16(false, a, false, b, (short)0, c,
                                                 false, false);
}

// ---------------------------------------------------------------------------
// Kernel 0: one-time f32 -> bf16 conversion of activations and weights.
// grid = (2048, 1, 7), block = 256; each thread converts 8 elements.
// z: 0..2 = X_Q/X_K/X_V (4Mi), 3..6 = W_Q/W_K/W_V/W_out (1Mi).
// ---------------------------------------------------------------------------
__global__ __launch_bounds__(256) void mha_cvt_kernel(
    const float* __restrict__ s0, const float* __restrict__ s1,
    const float* __restrict__ s2, const float* __restrict__ s3,
    const float* __restrict__ s4, const float* __restrict__ s5,
    const float* __restrict__ s6,
    __bf16* __restrict__ d0, __bf16* __restrict__ d1, __bf16* __restrict__ d2,
    __bf16* __restrict__ d3, __bf16* __restrict__ d4, __bf16* __restrict__ d5,
    __bf16* __restrict__ d6) {
  const int z = blockIdx.z;
  const float* src;
  __bf16* dst;
  size_t n;
  switch (z) {
    case 0: src = s0; dst = d0; n = XSZ; break;
    case 1: src = s1; dst = d1; n = XSZ; break;
    case 2: src = s2; dst = d2; n = XSZ; break;
    case 3: src = s3; dst = d3; n = WSZ; break;
    case 4: src = s4; dst = d4; n = WSZ; break;
    case 5: src = s5; dst = d5; n = WSZ; break;
    default: src = s6; dst = d6; n = WSZ; break;
  }
  const size_t i = ((size_t)blockIdx.x * 256 + threadIdx.x) * 8;
  if (i >= n) return;
  float4 f0 = *(const float4*)(src + i);
  float4 f1 = *(const float4*)(src + i + 4);
  union { uint4 q; __bf16 e[8]; } u;
  u.e[0] = f2bf(f0.x); u.e[1] = f2bf(f0.y); u.e[2] = f2bf(f0.z); u.e[3] = f2bf(f0.w);
  u.e[4] = f2bf(f1.x); u.e[5] = f2bf(f1.y); u.e[6] = f2bf(f1.z); u.e[7] = f2bf(f1.w);
  *(uint4*)(dst + i) = u.q;
}

// ---------------------------------------------------------------------------
// Kernel 1: Q/K/V projections (all-bf16 GEMM). grid = (M/32, N/256, 3),
// block = 128 (4 waves). Wave computes a 32x64 tile of Y = X @ W^T + b.
// z=0 -> Q (bf16 [b][h][l][64]); z=1 -> K (same); z=2 -> V transposed [b][h][d][L].
// ---------------------------------------------------------------------------
__global__ __launch_bounds__(128) void mha_proj_kernel(
    const __bf16* __restrict__ Xq, const __bf16* __restrict__ Xk,
    const __bf16* __restrict__ Xv,
    const __bf16* __restrict__ Wq, const __bf16* __restrict__ Wk,
    const __bf16* __restrict__ Wv,
    const float* __restrict__ bq, const float* __restrict__ bk,
    const float* __restrict__ bv,
    __bf16* __restrict__ Qb, __bf16* __restrict__ Kb, __bf16* __restrict__ Vt) {
  const int z = blockIdx.z;
  const __bf16* X    = (z == 0) ? Xq : (z == 1) ? Xk : Xv;
  const __bf16* W    = (z == 0) ? Wq : (z == 1) ? Wk : Wv;
  const float*  bias = (z == 0) ? bq : (z == 1) ? bk : bv;

  const int wave = threadIdx.x >> 5;
  const int lane = threadIdx.x & 31;
  const int lm = lane & 15, hi = lane >> 4;
  const int mbase = blockIdx.x * 32;
  const int nbase = blockIdx.y * 256 + wave * 64;

  const __bf16* xr0 = X + (size_t)(mbase + lm) * D_MODEL;
  const __bf16* xr1 = X + (size_t)(mbase + 16 + lm) * D_MODEL;
  const __bf16* wr0 = W + (size_t)(nbase +  0 + lm) * D_MODEL;
  const __bf16* wr1 = W + (size_t)(nbase + 16 + lm) * D_MODEL;
  const __bf16* wr2 = W + (size_t)(nbase + 32 + lm) * D_MODEL;
  const __bf16* wr3 = W + (size_t)(nbase + 48 + lm) * D_MODEL;

  v8f c[8] = {};  // [mt*4 + nt]
  for (int kk = 0; kk < D_MODEL; kk += 32) {
    const int ka = kk + hi * 8;
    const int kb = kk + hi * 8;
    v16bf a0 = load_a_bf16(xr0, ka);
    v16bf a1 = load_a_bf16(xr1, ka);
    v16bf b0 = load_b_bf16(wr0, kb);
    v16bf b1 = load_b_bf16(wr1, kb);
    v16bf b2 = load_b_bf16(wr2, kb);
    v16bf b3 = load_b_bf16(wr3, kb);
    c[0] = wmma_bf16(a0, b0, c[0]);
    c[1] = wmma_bf16(a0, b1, c[1]);
    c[2] = wmma_bf16(a0, b2, c[2]);
    c[3] = wmma_bf16(a0, b3, c[3]);
    c[4] = wmma_bf16(a1, b0, c[4]);
    c[5] = wmma_bf16(a1, b1, c[5]);
    c[6] = wmma_bf16(a1, b2, c[6]);
    c[7] = wmma_bf16(a1, b3, c[7]);
  }

#pragma unroll
  for (int mt = 0; mt < 2; ++mt) {
#pragma unroll
    for (int nt = 0; nt < 4; ++nt) {
      const int n  = nbase + nt * 16 + lm;
      const float bb = bias[n];
      const int hh = n >> 6, d = n & 63;
#pragma unroll
      for (int j = 0; j < 8; ++j) {
        const int m = mbase + mt * 16 + j + hi * 8;  // row in 0..4095
        const int bidx = m >> 11, l = m & 2047;
        const float y = c[mt * 4 + nt][j] + bb;
        if (z == 0) {
          Qb[(((size_t)(bidx * NUM_HEADS + hh) * SEQ) + l) * DK + d] = f2bf(y);
        } else if (z == 1) {
          Kb[(((size_t)(bidx * NUM_HEADS + hh) * SEQ) + l) * DK + d] = f2bf(y);
        } else {
          Vt[(((size_t)(bidx * NUM_HEADS + hh) * DK) + d) * SEQ + l] = f2bf(y);
        }
      }
    }
  }
}

// ---------------------------------------------------------------------------
// Kernel 2: flash attention. One wave per (b, h, 16-row q-tile).
// grid = B*H*(L/16)/4 blocks of 128 threads (4 independent waves per block).
// ---------------------------------------------------------------------------
__global__ __launch_bounds__(128) void mha_attn_kernel(
    const unsigned char* __restrict__ mask,
    const __bf16* __restrict__ Qb, const __bf16* __restrict__ Kb,
    const __bf16* __restrict__ Vt, __bf16* __restrict__ Ob) {
  __shared__ __bf16 Pl[4][16][32];  // per-wave P-tile staging (C-layout -> A-layout)

  const int wave = threadIdx.x >> 5;
  const int lane = threadIdx.x & 31;
  const int lm = lane & 15, hi = lane >> 4;
  const int gw = blockIdx.x * 4 + wave;
  const int qt = gw & 127;
  const int bh = gw >> 7;          // 0..31
  const int b = bh >> 4, h = bh & 15;
  const int qbase = qt * 16;

  const __bf16* qrow = Qb + ((size_t)bh * SEQ + qbase + lm) * DK;
  const v16bf aq0 = load_a_bf16(qrow, 0 + hi * 8);
  const v16bf aq1 = load_a_bf16(qrow, 32 + hi * 8);

  v8f oc0 = {}, oc1 = {}, oc2 = {}, oc3 = {};
  v8f mrow, lrow;
#pragma unroll
  for (int j = 0; j < 8; ++j) { mrow[j] = -3.0e38f; lrow[j] = 0.0f; }

  const unsigned char* mrowp = mask + ((size_t)b * SEQ + qbase) * SEQ;
  const __bf16* vbase = Vt + (size_t)bh * DK * SEQ;

  for (int kb = 0; kb < SEQ; kb += 32) {
    // ---- S = (Q K^T) * 1/8 with mask, for 32 keys (two 16-col tiles) ----
    v8f s[2];
#pragma unroll
    for (int sub = 0; sub < 2; ++sub) {
      const __bf16* krow = Kb + ((size_t)bh * SEQ + kb + sub * 16 + lm) * DK;
      v16bf bk0 = load_b_bf16(krow, 0 + hi * 8);
      v16bf bk1 = load_b_bf16(krow, 32 + hi * 8);
      v8f t = {};
      t = wmma_bf16(aq0, bk0, t);
      t = wmma_bf16(aq1, bk1, t);
      const int keyc = kb + sub * 16 + lm;
      v8f r;
#pragma unroll
      for (int j = 0; j < 8; ++j) {
        const int row = j + hi * 8;
        float v = t[j] * 0.125f;  // 1/sqrt(64)
        if (mrowp[(size_t)row * SEQ + keyc]) v = -10000.0f;
        r[j] = v;
      }
      s[sub] = r;
    }

    // ---- online softmax update (row reductions across 16-lane halves) ----
    v8f newm;
#pragma unroll
    for (int j = 0; j < 8; ++j) {
      float t = fmaxf(s[0][j], s[1][j]);
      for (int off = 1; off < 16; off <<= 1) t = fmaxf(t, __shfl_xor(t, off, 32));
      newm[j] = fmaxf(mrow[j], t);
    }
    v8f p0, p1, pf;
#pragma unroll
    for (int j = 0; j < 8; ++j) {
      p0[j] = __expf(s[0][j] - newm[j]);
      p1[j] = __expf(s[1][j] - newm[j]);
      pf[j] = __expf(mrow[j] - newm[j]);
    }
#pragma unroll
    for (int j = 0; j < 8; ++j) {
      float t = p0[j] + p1[j];
      for (int off = 1; off < 16; off <<= 1) t += __shfl_xor(t, off, 32);
      lrow[j] = lrow[j] * pf[j] + t;
      mrow[j] = newm[j];
    }
#pragma unroll
    for (int j = 0; j < 8; ++j) {
      oc0[j] *= pf[j]; oc1[j] *= pf[j]; oc2[j] *= pf[j]; oc3[j] *= pf[j];
    }

    // ---- P (16x32) through LDS: C-layout -> A-layout ----
#pragma unroll
    for (int j = 0; j < 8; ++j) {
      const int row = j + hi * 8;
      Pl[wave][row][lm]      = f2bf(p0[j]);
      Pl[wave][row][lm + 16] = f2bf(p1[j]);
    }
    const v16bf pa = load_a_bf16(&Pl[wave][lm][0], hi * 8);

    // ---- O += P V (V stored transposed: B-fragments are contiguous keys) ----
    const int kv = kb + hi * 8;
    oc0 = wmma_bf16(pa, load_b_bf16(vbase + (size_t)( 0 + lm) * SEQ, kv), oc0);
    oc1 = wmma_bf16(pa, load_b_bf16(vbase + (size_t)(16 + lm) * SEQ, kv), oc1);
    oc2 = wmma_bf16(pa, load_b_bf16(vbase + (size_t)(32 + lm) * SEQ, kv), oc2);
    oc3 = wmma_bf16(pa, load_b_bf16(vbase + (size_t)(48 + lm) * SEQ, kv), oc3);
  }

  // ---- finalize: O /= l, write (B, L, H*64) bf16 for the out projection ----
  v8f inv;
#pragma unroll
  for (int j = 0; j < 8; ++j) inv[j] = 1.0f / lrow[j];
  __bf16* obase = Ob + ((size_t)b * SEQ + qbase) * (NUM_HEADS * DK) + h * DK;
#pragma unroll
  for (int j = 0; j < 8; ++j) {
    const int row = j + hi * 8;
    __bf16* orow = obase + (size_t)row * (NUM_HEADS * DK);
    orow[ 0 + lm] = f2bf(oc0[j] * inv[j]);
    orow[16 + lm] = f2bf(oc1[j] * inv[j]);
    orow[32 + lm] = f2bf(oc2[j] * inv[j]);
    orow[48 + lm] = f2bf(oc3[j] * inv[j]);
  }
}

// ---------------------------------------------------------------------------
// Kernel 3: out = Ob @ W_out^T + b_out (fp32 epilogue to d_out).
// grid = (M/32, N/256), block = 128 (4 waves, 32x64 tile per wave).
// ---------------------------------------------------------------------------
__global__ __launch_bounds__(128) void mha_outproj_kernel(
    const __bf16* __restrict__ Ob, const __bf16* __restrict__ Wout,
    const float* __restrict__ bout, float* __restrict__ out) {
  const int wave = threadIdx.x >> 5;
  const int lane = threadIdx.x & 31;
  const int lm = lane & 15, hi = lane >> 4;
  const int mbase = blockIdx.x * 32;
  const int nbase = blockIdx.y * 256 + wave * 64;

  const __bf16* ar0 = Ob + (size_t)(mbase + lm) * D_MODEL;
  const __bf16* ar1 = Ob + (size_t)(mbase + 16 + lm) * D_MODEL;
  const __bf16* wr0 = Wout + (size_t)(nbase +  0 + lm) * D_MODEL;
  const __bf16* wr1 = Wout + (size_t)(nbase + 16 + lm) * D_MODEL;
  const __bf16* wr2 = Wout + (size_t)(nbase + 32 + lm) * D_MODEL;
  const __bf16* wr3 = Wout + (size_t)(nbase + 48 + lm) * D_MODEL;

  v8f c[8] = {};
  for (int kk = 0; kk < D_MODEL; kk += 32) {
    const int ka = kk + hi * 8;
    const int kb = kk + hi * 8;
    v16bf a0 = load_a_bf16(ar0, ka);
    v16bf a1 = load_a_bf16(ar1, ka);
    v16bf b0 = load_b_bf16(wr0, kb);
    v16bf b1 = load_b_bf16(wr1, kb);
    v16bf b2 = load_b_bf16(wr2, kb);
    v16bf b3 = load_b_bf16(wr3, kb);
    c[0] = wmma_bf16(a0, b0, c[0]);
    c[1] = wmma_bf16(a0, b1, c[1]);
    c[2] = wmma_bf16(a0, b2, c[2]);
    c[3] = wmma_bf16(a0, b3, c[3]);
    c[4] = wmma_bf16(a1, b0, c[4]);
    c[5] = wmma_bf16(a1, b1, c[5]);
    c[6] = wmma_bf16(a1, b2, c[6]);
    c[7] = wmma_bf16(a1, b3, c[7]);
  }

#pragma unroll
  for (int mt = 0; mt < 2; ++mt) {
#pragma unroll
    for (int nt = 0; nt < 4; ++nt) {
      const int n = nbase + nt * 16 + lm;
      const float bb = bout[n];
#pragma unroll
      for (int j = 0; j < 8; ++j) {
        const int m = mbase + mt * 16 + j + hi * 8;
        out[(size_t)m * D_MODEL + n] = c[mt * 4 + nt][j] + bb;
      }
    }
  }
}

// ---------------------------------------------------------------------------
extern "C" void kernel_launch(void* const* d_in, const int* in_sizes, int n_in,
                              void* d_out, int out_size, void* d_ws, size_t ws_size,
                              hipStream_t stream) {
  const float* inQ = (const float*)d_in[0];
  const float* inK = (const float*)d_in[1];
  const float* inV = (const float*)d_in[2];
  const unsigned char* mask = (const unsigned char*)d_in[3];  // jnp.bool_ = 1 byte
  const float* WQ = (const float*)d_in[4];
  const float* bQ = (const float*)d_in[5];
  const float* WK = (const float*)d_in[6];
  const float* bK = (const float*)d_in[7];
  const float* WV = (const float*)d_in[8];
  const float* bV = (const float*)d_in[9];
  const float* Wo = (const float*)d_in[10];
  const float* bo = (const float*)d_in[11];

  // Workspace (bf16 elems): 7 activation-size buffers + 4 weight-size = 64 MB.
  __bf16* ws = (__bf16*)d_ws;
  __bf16* Qb  = ws;                    // [b][h][l][64]
  __bf16* Kb  = ws + 1 * XSZ;          // [b][h][l][64]
  __bf16* Vt  = ws + 2 * XSZ;          // [b][h][64][L] (transposed)
  __bf16* Ob  = ws + 3 * XSZ;          // [b][l][h*64]
  __bf16* Xqb = ws + 4 * XSZ;
  __bf16* Xkb = ws + 5 * XSZ;
  __bf16* Xvb = ws + 6 * XSZ;
  __bf16* Wqb = ws + 7 * XSZ;
  __bf16* Wkb = Wqb + WSZ;
  __bf16* Wvb = Wkb + WSZ;
  __bf16* Wob = Wvb + WSZ;

  // 0) one-time f32 -> bf16 staging
  dim3 cb(256), cg((unsigned)(XSZ / 8 / 256), 1, 7);
  mha_cvt_kernel<<<cg, cb, 0, stream>>>(inQ, inK, inV, WQ, WK, WV, Wo,
                                        Xqb, Xkb, Xvb, Wqb, Wkb, Wvb, Wob);

  // 1) Q/K/V projections
  dim3 pb(128), pg(MROWS / 32, D_MODEL / 256, 3);
  mha_proj_kernel<<<pg, pb, 0, stream>>>(Xqb, Xkb, Xvb, Wqb, Wkb, Wvb,
                                         bQ, bK, bV, Qb, Kb, Vt);

  // 2) flash attention
  dim3 ab(128), ag(BATCH * NUM_HEADS * (SEQ / 16) / 4);
  mha_attn_kernel<<<ag, ab, 0, stream>>>(mask, Qb, Kb, Vt, Ob);

  // 3) output projection
  dim3 ob(128), og(MROWS / 32, D_MODEL / 256);
  mha_outproj_kernel<<<og, ob, 0, stream>>>(Ob, Wob, bo, (float*)d_out);
}
